// MultiheadAttention_39135742001772
// MI455X (gfx1250) — matrix operand order repaired and checked by reading, not compiled, not run on hardware
//
#include <hip/hip_runtime.h>
#include <math.h>

// ---------------- problem constants ----------------
#define TDIM 4096
#define SDIM 4096
#define BDIM 4
#define EDIM 1024
#define HDIM 16
#define DH   64
#define BH   (BDIM * HDIM)     // 64
#define MROW (TDIM * BDIM)     // 16384
#define HALF_PI 1.57079632679489662f
#define EPSV 1e-6f

typedef __bf16 bf16_t;
typedef __attribute__((ext_vector_type(8)))  __bf16 v8bf;
typedef __attribute__((ext_vector_type(16))) __bf16 v16bf;
typedef __attribute__((ext_vector_type(8)))  float  v8f;
typedef __attribute__((ext_vector_type(4)))  unsigned v4u;
typedef __attribute__((ext_vector_type(4)))  int    v4i;
typedef __attribute__((ext_vector_type(8)))  int    v8i;

#if __has_builtin(__builtin_amdgcn_tensor_load_to_lds) || __has_builtin(__builtin_amdgcn_tensor_load_to_lds_d2)
#define HAVE_TDM 1
#else
#define HAVE_TDM 0
#endif

// ---------------- small helpers ----------------
__device__ __forceinline__ float bf2f(bf16_t x) {
  unsigned u = ((unsigned)__builtin_bit_cast(unsigned short, x)) << 16;
  return __builtin_bit_cast(float, u);
}
__device__ __forceinline__ bf16_t f2bf(float x) {
  unsigned u = __builtin_bit_cast(unsigned, x);
  u += 0x7FFFu + ((u >> 16) & 1u);          // round-to-nearest-even
  unsigned short s = (unsigned short)(u >> 16);
  return __builtin_bit_cast(bf16_t, s);
}
__device__ __forceinline__ v8f vzero8() {
  v8f z;
#pragma unroll
  for (int e = 0; e < 8; ++e) z[e] = 0.0f;
  return z;
}

// A-fragment (16x32 bf16): lane = 16*hi + r holds row r, K = {hi*8..hi*8+7, 16+hi*8..16+hi*8+7}
__device__ __forceinline__ v16bf load_frag_a(const bf16_t* rowp, int hi) {
  const v8bf lo = *(const v8bf*)(rowp + hi * 8);
  const v8bf hh = *(const v8bf*)(rowp + hi * 8 + 16);
  v16bf f;
#pragma unroll
  for (int j = 0; j < 8; ++j) { f[j] = lo[j]; f[j + 8] = hh[j]; }
  return f;
}
// B-fragment (32x16 bf16): lane = 16*hi + r holds column r, K = {hi*8..hi*8+7, 16+hi*8..16+hi*8+7}
__device__ __forceinline__ v16bf load_frag_b(const bf16_t* colp, int hi) {
  const v8bf lo = *(const v8bf*)(colp + hi * 8);
  const v8bf hh = *(const v8bf*)(colp + hi * 8 + 16);
  v16bf f;
#pragma unroll
  for (int j = 0; j < 8; ++j) { f[j] = lo[j]; f[j + 8] = hh[j]; }
  return f;
}
__device__ __forceinline__ v8f wmma_bf16(v16bf a, v16bf b, v8f c) {
  return __builtin_amdgcn_wmma_f32_16x16x32_bf16(
      /*neg_a=*/false, a, /*neg_b=*/false, b,
      /*c_mod=*/(short)0, c, /*reuse_a=*/false, /*reuse_b=*/false);
}

__device__ __forceinline__ void wait_tensorcnt0() {
#if __has_builtin(__builtin_amdgcn_s_wait_tensorcnt)
  __builtin_amdgcn_s_wait_tensorcnt(0);
#else
  asm volatile("s_wait_tensorcnt 0x0" ::: "memory");
#endif
}

#if HAVE_TDM
// Issue a TDM 2-D tile load: 64 rows x 32 bf16 cols from a [*,1024] bf16 tensor into LDS.
// LDS rows are padded by TDM (16B pad after each 64B row) -> row stride 40 elements (80B).
__device__ __forceinline__ void tdm_load_tile64x32(const bf16_t* gtile, unsigned lds_byte_off) {
  const unsigned long long ga = (unsigned long long)(uintptr_t)gtile;
  v4u g0;
  g0[0] = 1u;                                      // count=1, user descriptor
  g0[1] = lds_byte_off;                            // lds_addr
  g0[2] = (unsigned)(ga & 0xFFFFFFFFu);            // global_addr[31:0]
  g0[3] = (unsigned)((ga >> 32) & 0x1FFFFFFu) | (2u << 30);  // global_addr[56:32] | type=2
  v8i g1;
  // data_size=2B (1<<16), pad_enable (1<<20), pad_interval: 16 DWORDs (3<<22), pad_amount: 4 DWORDs (3<<25)
  g1[0] = (int)((1u << 16) | (1u << 20) | (3u << 22) | (3u << 25));
  g1[1] = (int)(1024u << 16);                      // tensor_dim0 = 1024 (bits [79:48], low part)
  g1[2] = (int)(16384u << 16);                     // tensor_dim0 hi=0 | tensor_dim1 = 16384 (low part)
  g1[3] = (int)(32u << 16);                        // tensor_dim1 hi=0 | tile_dim0 = 32
  g1[4] = 64;                                      // tile_dim1 = 64, tile_dim2 = 0
  g1[5] = 1024;                                    // tensor_dim0_stride = 1024 (elements)
  g1[6] = 0;                                       // stride hi | tensor_dim1_stride lo
  g1[7] = 0;
#if __has_builtin(__builtin_amdgcn_tensor_load_to_lds_d2)
  __builtin_amdgcn_tensor_load_to_lds_d2(g0, g1, 0);
#elif defined(__clang_major__) && (__clang_major__ >= 23)
  v4i zz = {0, 0, 0, 0};
  v8i z8 = {0, 0, 0, 0, 0, 0, 0, 0};
  __builtin_amdgcn_tensor_load_to_lds(g0, g1, zz, zz, z8, 0);
#else
  v4i zz = {0, 0, 0, 0};
  __builtin_amdgcn_tensor_load_to_lds(g0, g1, zz, zz, 0);
#endif
}
#endif  // HAVE_TDM

// ---------------- fp32 -> bf16 convert ----------------
__global__ void cvt_f32_bf16(const float* __restrict__ src,
                             bf16_t* __restrict__ dst, int n) {
  int i = blockIdx.x * blockDim.x + threadIdx.x;
  if (i < n) dst[i] = f2bf(src[i]);
}

// ---------------- projection GEMM: out = A[M,K] * W[N,K]^T + bias, fused epilogue ----------------
// A tile (64 rows x 32 k, shared by all 8 waves of the block) is staged into LDS by the
// Tensor Data Mover, double-buffered and overlapped with WMMA compute. W fragments are
// read directly from global (weights are L2-resident).
// mode 0: q  -> scale, relu, * sin/cos((t+1)/T)   writes out0(bf16 sin), out1(bf16 cos)
// mode 1: k  -> mask,  relu, * sin/cos((s+1)/S)   writes out0, out1
// mode 2: v  -> bf16                               writes out0
// mode 3: o  -> f32                                writes outf
#define APAD 40   // LDS row stride in elements (32 data + 8 pad), 80B = 16B aligned
__global__ void __launch_bounds__(256) proj_gemm_kernel(
    const bf16_t* __restrict__ A, const bf16_t* __restrict__ W,
    const float* __restrict__ bias, int mode, float scale,
    const unsigned char* __restrict__ mask,
    bf16_t* __restrict__ out0, bf16_t* __restrict__ out1,
    float* __restrict__ outf)
{
  const int lane = threadIdx.x & 31;
  const int r    = lane & 15;
  const int hi   = lane >> 4;
  const int widx = threadIdx.x >> 5;                 // 0..7
  const int wave = blockIdx.x * 8 + widx;
  const int mt = (blockIdx.x * 8) >> 4;              // block-uniform (8 waves share A rows)
  const int nt = wave & 15;
  const int row0 = mt * 64;
  const int col0 = nt * 64;

  const bf16_t* Wcol[4];
#pragma unroll
  for (int i = 0; i < 4; ++i)
    Wcol[i] = W + (size_t)(col0 + i * 16 + r) * EDIM;

  v8f acc[4][4];
#pragma unroll
  for (int i = 0; i < 4; ++i)
#pragma unroll
    for (int j = 0; j < 4; ++j) acc[i][j] = vzero8();

#if HAVE_TDM
  __shared__ __attribute__((aligned(16))) bf16_t ldsA[2][64 * APAD];
  const bf16_t* Atile0 = A + (size_t)row0 * EDIM;
  const bool tdmWave = (widx == 0);
  if (tdmWave) {
    tdm_load_tile64x32(Atile0, (unsigned)(uintptr_t)&ldsA[0][0]);
    wait_tensorcnt0();
  }
  __syncthreads();
  for (int k0 = 0; k0 < EDIM; k0 += 32) {
    const int cur = (k0 >> 5) & 1;
    if (tdmWave && (k0 + 32 < EDIM))
      tdm_load_tile64x32(Atile0 + k0 + 32, (unsigned)(uintptr_t)&ldsA[cur ^ 1][0]);
    if (k0 + 128 < EDIM) {
      __builtin_prefetch(Wcol[0] + k0 + 128, 0, 3);  // global_prefetch_b8 (near)
      __builtin_prefetch(Wcol[2] + k0 + 128, 0, 3);
    }
    v16bf af[4], bf[4];
#pragma unroll
    for (int i = 0; i < 4; ++i) af[i] = load_frag_a(&ldsA[cur][(i * 16 + r) * APAD], hi);
#pragma unroll
    for (int j = 0; j < 4; ++j) bf[j] = load_frag_b(Wcol[j] + k0, hi);
#pragma unroll
    for (int i = 0; i < 4; ++i)
#pragma unroll
      for (int j = 0; j < 4; ++j)
        acc[i][j] = wmma_bf16(af[i], bf[j], acc[i][j]);
    if (tdmWave) wait_tensorcnt0();                  // next buffer landed
    __syncthreads();
  }
#else
  const bf16_t* Arow[4];
#pragma unroll
  for (int i = 0; i < 4; ++i)
    Arow[i] = A + (size_t)(row0 + i * 16 + r) * EDIM;
  for (int k0 = 0; k0 < EDIM; k0 += 32) {
    if (k0 + 128 < EDIM) {
      __builtin_prefetch(Arow[0] + k0 + 128, 0, 3);
      __builtin_prefetch(Arow[2] + k0 + 128, 0, 3);
    }
    v16bf af[4], bf[4];
#pragma unroll
    for (int i = 0; i < 4; ++i) af[i] = load_frag_a(Arow[i] + k0, hi);
#pragma unroll
    for (int j = 0; j < 4; ++j) bf[j] = load_frag_b(Wcol[j] + k0, hi);
#pragma unroll
    for (int i = 0; i < 4; ++i)
#pragma unroll
      for (int j = 0; j < 4; ++j)
        acc[i][j] = wmma_bf16(af[i], bf[j], acc[i][j]);
  }
#endif

  // epilogue: C element (vgpr e, lane) -> M = e + 8*hi, N = r
#pragma unroll
  for (int i = 0; i < 4; ++i) {
#pragma unroll
    for (int e = 0; e < 8; ++e) {
      const int gm = row0 + i * 16 + hi * 8 + e;      // row = l*BDIM + b
      float sf = 0.0f, cf = 0.0f;
      int masked = 0;
      if (mode == 0) {
        const int t = gm >> 2;                        // BDIM == 4
        const float ang = HALF_PI * (float)(t + 1) * (1.0f / (float)TDIM);
        sf = sinf(ang); cf = cosf(ang);
      } else if (mode == 1) {
        const int s = gm >> 2;
        const int bb = gm & 3;
        masked = mask ? (int)mask[bb * SDIM + s] : 0;
        const float ang = HALF_PI * (float)(s + 1) * (1.0f / (float)SDIM);
        sf = sinf(ang); cf = cosf(ang);
      }
#pragma unroll
      for (int j = 0; j < 4; ++j) {
        const int gn = col0 + j * 16 + r;
        float v = acc[i][j][e] + bias[gn];
        const size_t idx = (size_t)gm * EDIM + gn;
        if (mode == 0) {
          v *= scale; v = v > 0.0f ? v : 0.0f;
          out0[idx] = f2bf(v * sf);
          out1[idx] = f2bf(v * cf);
        } else if (mode == 1) {
          if (masked) v = 0.0f;
          v = v > 0.0f ? v : 0.0f;
          out0[idx] = f2bf(v * sf);
          out1[idx] = f2bf(v * cf);
        } else if (mode == 2) {
          out0[idx] = f2bf(v);
        } else {
          outf[idx] = v;
        }
      }
    }
  }
}

// ---------------- kv_phi = k_phi^T @ v per (b,h); writes kv TRANSPOSED [bh][d2][d1] bf16 ----------------
#define CHS  32
#define PADS 40   // padded LDS row stride (elements); 80B rows keep 16B alignment
__global__ void __launch_bounds__(512) kv_kernel(
    const bf16_t* __restrict__ k_sin, const bf16_t* __restrict__ k_cos,
    const bf16_t* __restrict__ vmat,
    bf16_t* __restrict__ kvT_sin, bf16_t* __restrict__ kvT_cos)
{
  __shared__ __attribute__((aligned(16))) bf16_t lsk[64 * PADS];  // k_sin^T chunk
  __shared__ __attribute__((aligned(16))) bf16_t lck[64 * PADS];  // k_cos^T chunk
  __shared__ __attribute__((aligned(16))) bf16_t lvv[64 * PADS];  // v^T chunk
  const int bh = blockIdx.x;
  const int b  = bh / HDIM;
  const int h  = bh % HDIM;
  const int tid  = threadIdx.x;
  const int lane = tid & 31;
  const int r    = lane & 15;
  const int hi   = lane >> 4;
  const int wave = tid >> 5;        // 0..15
  const int d1t  = wave >> 2;       // 0..3
  const int d2t  = wave & 3;        // 0..3

  v8f accS = vzero8();
  v8f accC = vzero8();
  const size_t headoff = (size_t)b * EDIM + (size_t)h * DH;

  for (int s0 = 0; s0 < SDIM; s0 += CHS) {
    // cooperative transposed staging: lds[d][s_local] = x[s0+s_local][d]
#pragma unroll
    for (int it = 0; it < 4; ++it) {
      const int idx = tid + it * 512;           // 0..2047 covers 32x64
      const int sl  = idx >> 6;
      const int dc  = idx & 63;
      const size_t g = (size_t)(s0 + sl) * (BDIM * EDIM) + headoff + dc;
      const int l = dc * PADS + sl;
      lsk[l] = k_sin[g];
      lck[l] = k_cos[g];
      lvv[l] = vmat[g];
    }
    __syncthreads();
    const v16bf aS = load_frag_a(&lsk[(d1t * 16 + r) * PADS], hi);
    const v16bf aC = load_frag_a(&lck[(d1t * 16 + r) * PADS], hi);
    const v16bf bV = load_frag_b(&lvv[(d2t * 16 + r) * PADS], hi);
    accS = wmma_bf16(aS, bV, accS);
    accC = wmma_bf16(aC, bV, accC);
    __syncthreads();
  }
#pragma unroll
  for (int e = 0; e < 8; ++e) {
    const int gd1 = d1t * 16 + hi * 8 + e;
    const int gd2 = d2t * 16 + r;
    const size_t o = ((size_t)bh * DH + gd2) * DH + gd1;   // transposed store
    kvT_sin[o] = f2bf(accS[e]);
    kvT_cos[o] = f2bf(accC[e]);
  }
}

// ---------------- ksum over S per (bh, d) ----------------
__global__ void ksum_kernel(const bf16_t* __restrict__ k_sin,
                            const bf16_t* __restrict__ k_cos,
                            float* __restrict__ ksum_sin,
                            float* __restrict__ ksum_cos)
{
  const int bh = blockIdx.x;
  const int d  = threadIdx.x;
  const int b = bh / HDIM, h = bh % HDIM;
  const size_t base = (size_t)b * EDIM + (size_t)h * DH + d;
  float ss = 0.0f, sc = 0.0f;
  for (int s = 0; s < SDIM; ++s) {
    const size_t g = base + (size_t)s * (BDIM * EDIM);
    ss += bf2f(k_sin[g]);
    sc += bf2f(k_cos[g]);
  }
  ksum_sin[bh * DH + d] = ss;
  ksum_cos[bh * DH + d] = sc;
}

// ---------------- denom[bh][t] = max(q_sin.ksum_sin + q_cos.ksum_cos, eps) ----------------
__global__ void denom_kernel(const bf16_t* __restrict__ q_sin,
                             const bf16_t* __restrict__ q_cos,
                             const float* __restrict__ ksum_sin,
                             const float* __restrict__ ksum_cos,
                             float* __restrict__ denom)
{
  const int gid = blockIdx.x * blockDim.x + threadIdx.x;  // bh*T + t
  const int bh = gid >> 12;      // T == 4096
  const int t  = gid & 4095;
  const int b = bh / HDIM, h = bh % HDIM;
  const size_t base = (size_t)t * (BDIM * EDIM) + (size_t)b * EDIM + (size_t)h * DH;
  const float* ks = ksum_sin + bh * DH;
  const float* kc = ksum_cos + bh * DH;
  float acc = 0.0f;
#pragma unroll 8
  for (int d = 0; d < DH; ++d)
    acc += bf2f(q_sin[base + d]) * ks[d] + bf2f(q_cos[base + d]) * kc[d];
  denom[gid] = fmaxf(acc, EPSV);
}

// ---------------- num = q_sin@kv_sin + q_cos@kv_cos, fused /denom ----------------
__global__ void __launch_bounds__(256) num_kernel(
    const bf16_t* __restrict__ q_sin, const bf16_t* __restrict__ q_cos,
    const bf16_t* __restrict__ kvT_sin, const bf16_t* __restrict__ kvT_cos,
    const float* __restrict__ denom, bf16_t* __restrict__ attn)
{
  const int lane = threadIdx.x & 31;
  const int r    = lane & 15;
  const int hi   = lane >> 4;
  const int wave = blockIdx.x * (blockDim.x >> 5) + (threadIdx.x >> 5);
  const int bh = wave >> 8;          // 256 t-tiles per bh
  const int tt = wave & 255;
  const int t0 = tt * 16;
  const int b = bh / HDIM, h = bh % HDIM;
  const size_t qbase  = ((size_t)(t0 + r) * BDIM + b) * EDIM + (size_t)h * DH;
  const size_t kvbase = (size_t)bh * (DH * DH);

  v8f acc[4];
#pragma unroll
  for (int j = 0; j < 4; ++j) acc[j] = vzero8();

#pragma unroll
  for (int ks = 0; ks < 2; ++ks) {   // K = 64 = 2 x 32
    const v16bf aS = load_frag_a(q_sin + qbase + ks * 32, hi);
    const v16bf aC = load_frag_a(q_cos + qbase + ks * 32, hi);
#pragma unroll
    for (int j = 0; j < 4; ++j) {
      const v16bf bS = load_frag_b(kvT_sin + kvbase + (size_t)(j * 16 + r) * DH + ks * 32, hi);
      const v16bf bC = load_frag_b(kvT_cos + kvbase + (size_t)(j * 16 + r) * DH + ks * 32, hi);
      acc[j] = wmma_bf16(aS, bS, acc[j]);
      acc[j] = wmma_bf16(aC, bC, acc[j]);
    }
  }
#pragma unroll
  for (int e = 0; e < 8; ++e) {
    const int gt = t0 + hi * 8 + e;
    const float rdn = 1.0f / denom[bh * TDIM + gt];
    const size_t obase = ((size_t)gt * BDIM + b) * EDIM + (size_t)h * DH;
#pragma unroll
    for (int j = 0; j < 4; ++j)
      attn[obase + j * 16 + r] = f2bf(acc[j][e] * rdn);
  }
}

// ---------------- host launcher ----------------
extern "C" void kernel_launch(void* const* d_in, const int* in_sizes, int n_in,
                              void* d_out, int out_size, void* d_ws, size_t ws_size,
                              hipStream_t stream)
{
  const float* query = (const float*)d_in[0];
  const float* key   = (const float*)d_in[1];
  const unsigned char* mask = (const unsigned char*)d_in[2];
  const float* Wq = (const float*)d_in[3];
  const float* bq = (const float*)d_in[4];
  const float* Wk = (const float*)d_in[5];
  const float* bk = (const float*)d_in[6];
  const float* Wv = (const float*)d_in[7];
  const float* bv = (const float*)d_in[8];
  const float* Wo = (const float*)d_in[9];
  const float* bo = (const float*)d_in[10];
  float* out = (float*)d_out;

  char* ws = (char*)d_ws;
  size_t off = 0;
  auto alloc = [&](size_t bytes) -> char* {
    char* p = ws + off;
    off = (off + bytes + 255) & ~(size_t)255;
    return p;
  };
  const size_t ME = (size_t)MROW * EDIM;          // 16.7M elements
  bf16_t* qbf   = (bf16_t*)alloc(ME * 2);
  bf16_t* kbf   = (bf16_t*)alloc(ME * 2);
  bf16_t* wqb   = (bf16_t*)alloc((size_t)EDIM * EDIM * 2);
  bf16_t* wkb   = (bf16_t*)alloc((size_t)EDIM * EDIM * 2);
  bf16_t* wvb   = (bf16_t*)alloc((size_t)EDIM * EDIM * 2);
  bf16_t* wob   = (bf16_t*)alloc((size_t)EDIM * EDIM * 2);
  bf16_t* q_sin = (bf16_t*)alloc(ME * 2);
  bf16_t* q_cos = (bf16_t*)alloc(ME * 2);
  bf16_t* k_sin = (bf16_t*)alloc(ME * 2);
  bf16_t* k_cos = (bf16_t*)alloc(ME * 2);
  bf16_t* vbf   = (bf16_t*)alloc(ME * 2);
  bf16_t* attn  = (bf16_t*)alloc(ME * 2);
  bf16_t* kvS   = (bf16_t*)alloc((size_t)BH * DH * DH * 2);
  bf16_t* kvC   = (bf16_t*)alloc((size_t)BH * DH * DH * 2);
  float*  ksS   = (float*)alloc((size_t)BH * DH * 4);
  float*  ksC   = (float*)alloc((size_t)BH * DH * 4);
  float*  den   = (float*)alloc((size_t)BH * TDIM * 4);

  // fp32 -> bf16 conversions
  cvt_f32_bf16<<<(int)(ME / 256), 256, 0, stream>>>(query, qbf, (int)ME);
  cvt_f32_bf16<<<(int)(ME / 256), 256, 0, stream>>>(key,   kbf, (int)ME);
  const int WN = EDIM * EDIM;
  cvt_f32_bf16<<<WN / 256, 256, 0, stream>>>(Wq, wqb, WN);
  cvt_f32_bf16<<<WN / 256, 256, 0, stream>>>(Wk, wkb, WN);
  cvt_f32_bf16<<<WN / 256, 256, 0, stream>>>(Wv, wvb, WN);
  cvt_f32_bf16<<<WN / 256, 256, 0, stream>>>(Wo, wob, WN);

  // projections (each wave owns a 64x64 tile; 8 waves/block share the A tile via TDM)
  const int gemmBlocks = (MROW / 64) * (EDIM / 64) / 8;   // 512
  const float scaling = 0.125f;                            // d^-0.5 = 64^-0.5
  proj_gemm_kernel<<<gemmBlocks, 256, 0, stream>>>(qbf, wqb, bq, 0, scaling, nullptr, q_sin, q_cos, nullptr);
  proj_gemm_kernel<<<gemmBlocks, 256, 0, stream>>>(kbf, wkb, bk, 1, 1.0f,    mask,    k_sin, k_cos, nullptr);
  proj_gemm_kernel<<<gemmBlocks, 256, 0, stream>>>(kbf, wvb, bv, 2, 1.0f,    nullptr, vbf,   nullptr, nullptr);

  // linear-attention core
  kv_kernel<<<BH, 512, 0, stream>>>(k_sin, k_cos, vbf, kvS, kvC);
  ksum_kernel<<<BH, DH, 0, stream>>>(k_sin, k_cos, ksS, ksC);
  denom_kernel<<<(BH * TDIM) / 256, 256, 0, stream>>>(q_sin, q_cos, ksS, ksC, den);
  num_kernel<<<(BH * (TDIM / 16)) / 8, 256, 0, stream>>>(q_sin, q_cos, kvS, kvC, den, attn);

  // output projection -> f32 d_out
  proj_gemm_kernel<<<gemmBlocks, 256, 0, stream>>>(attn, wob, bo, 3, 1.0f, nullptr, nullptr, nullptr, out);
}
